// AttentionLayer_17789754540266
// MI455X (gfx1250) — matrix operand order repaired and checked by reading, not compiled, not run on hardware
//
#include <hip/hip_runtime.h>

// ---------------------------------------------------------------------------
// AttentionLayer for MI455X (gfx1250): bf16 WMMA (f32 accumulate) everywhere,
// TDM (tensor_load_to_lds) staging for contiguous LDS tiles.
// B=8, S=2048, D_IN=D_ATT=512.
// ---------------------------------------------------------------------------

#define BATCH 8
#define SEQ   2048
#define DIM   512
#define BS    (BATCH * SEQ)          // 16384 rows

typedef __bf16 v16bf __attribute__((ext_vector_type(16)));
typedef float  v8f   __attribute__((ext_vector_type(8)));
typedef unsigned u32x4 __attribute__((ext_vector_type(4)));
typedef int      i32x4 __attribute__((ext_vector_type(4)));
typedef int      i32x8 __attribute__((ext_vector_type(8)));

__device__ __forceinline__ __bf16 f2bf(float f) {
  union { float f; unsigned u; } t; t.f = f;
  unsigned r = t.u + 0x7FFFu + ((t.u >> 16) & 1u);   // round-to-nearest-even
  union { unsigned short s; __bf16 b; } o;
  o.s = (unsigned short)(r >> 16);
  return o.b;
}

// A fragment (16x32, bf16), source stored row-major [M][K], row stride ld.
// ISA 7.12.2: lanes 0-15 -> M=lane, K in {0..7, 16..23};
//             lanes 16-31 -> K in {8..15, 24..31}. Two contiguous 16B runs.
__device__ __forceinline__ v16bf load_a_frag(const __bf16* tile, int ld) {
  int lane = threadIdx.x & 31;
  int m = lane & 15, kg = lane >> 4;
  const __bf16* p = tile + m * ld + kg * 8;
  union { v16bf v; uint4 u[2]; } t;
  t.u[0] = *(const uint4*)(p);
  t.u[1] = *(const uint4*)(p + 16);
  return t.v;
}

// B fragment (32x16, bf16), source stored TRANSPOSED [N][K], row stride ld.
__device__ __forceinline__ v16bf load_b_frag(const __bf16* tileT, int ld) {
  int lane = threadIdx.x & 31;
  int n = lane & 15, kg = lane >> 4;
  const __bf16* p = tileT + n * ld + kg * 8;
  union { v16bf v; uint4 u[2]; } t;
  t.u[0] = *(const uint4*)(p);
  t.u[1] = *(const uint4*)(p + 16);
  return t.v;
}

__device__ __forceinline__ v8f wmma_bf16(v16bf a, v16bf b, v8f c) {
  return __builtin_amdgcn_wmma_f32_16x16x32_bf16(false, a, false, b,
                                                 (short)0, c, false, false);
}

__device__ __forceinline__ v8f v8f_zero() {
  v8f z = {0.f, 0.f, 0.f, 0.f, 0.f, 0.f, 0.f, 0.f};
  return z;
}

// ---------------------------------------------------------------------------
// TDM staging of a contiguous run of n bf16 elements (n <= 16384) into LDS.
// Issued by the whole workgroup via wave 0 only; caller must __syncthreads()
// after. Falls back to a cooperative vector copy if the builtin is absent.
// ---------------------------------------------------------------------------
#if defined(__has_builtin)
#if __has_builtin(__builtin_amdgcn_tensor_load_to_lds)
#define HAVE_TDM 1
#endif
#endif

__device__ __forceinline__ void stage_tile_contig(void* lds_dst,
                                                  const void* gsrc,
                                                  unsigned n_bf16) {
#ifdef HAVE_TDM
  if (threadIdx.x < 32) {
    unsigned lds = (unsigned)(unsigned long long)lds_dst;
    unsigned long long ga = (unsigned long long)gsrc;
    // ---- D# group 0: count=1 | lds_addr | global_addr(57b) | type=2 ----
    u32x4 g0;
    g0[0] = 1u;                                         // count=1, user desc
    g0[1] = lds;                                        // lds_addr (bytes)
    g0[2] = (unsigned)(ga & 0xFFFFFFFFull);             // global_addr[31:0]
    g0[3] = (unsigned)((ga >> 32) & 0x1FFFFFFull)       // global_addr[56:32]
          | (2u << 30);                                 // type = 2 ("image")
    // ---- D# group 1: data_size=2B, 1-line tile of n elements ----
    i32x8 g1;
    g1[0] = 0x00010000;                                 // data_size=1 (2B)
    g1[1] = (int)((n_bf16 & 0xFFFFu) << 16);            // tensor_dim0[15:0]
    g1[2] = (int)(((n_bf16 >> 16) & 0xFFFFu)            // tensor_dim0[31:16]
          | (1u << 16));                                // tensor_dim1 = 1
    g1[3] = (int)((n_bf16 & 0xFFFFu) << 16);            // tile_dim0 = n
    g1[4] = 1;                                          // tile_dim1=1, dim2=0
    g1[5] = (int)n_bf16;                                // dim0_stride[31:0]
    g1[6] = 0;                                          // strides hi
    g1[7] = 0;
    i32x4 z4 = {0, 0, 0, 0};
#if __clang_major__ >= 23
    i32x8 z8 = {0, 0, 0, 0, 0, 0, 0, 0};
    __builtin_amdgcn_tensor_load_to_lds(g0, g1, z4, z4, z8, 0);
#else
    __builtin_amdgcn_tensor_load_to_lds(g0, g1, z4, z4, 0);
#endif
    __builtin_amdgcn_s_wait_tensorcnt(0);
  }
#else
  const uint4* src = (const uint4*)gsrc;
  uint4* dst = (uint4*)lds_dst;
  for (unsigned i = threadIdx.x; i < n_bf16 / 8; i += blockDim.x)
    dst[i] = src[i];
#endif
}

// ---------------------------------------------------------------------------
// Kernel 1: x fp32 -> bf16 (row-major).  One thread = 8 elements.
// ---------------------------------------------------------------------------
__global__ __launch_bounds__(256) void cvt_x_kernel(const float* __restrict__ x,
                                                    __bf16* __restrict__ xb) {
  int i = blockIdx.x * blockDim.x + threadIdx.x;
  const float4* s = (const float4*)x;
  float4 f0 = s[2 * i + 0];
  float4 f1 = s[2 * i + 1];
  union { __bf16 b[8]; uint4 u; } o;
  o.b[0] = f2bf(f0.x); o.b[1] = f2bf(f0.y); o.b[2] = f2bf(f0.z); o.b[3] = f2bf(f0.w);
  o.b[4] = f2bf(f1.x); o.b[5] = f2bf(f1.y); o.b[6] = f2bf(f1.z); o.b[7] = f2bf(f1.w);
  ((uint4*)xb)[i] = o.u;
}

// ---------------------------------------------------------------------------
// Kernel 2: Wq/Wk/Wv fp32 [d][e] -> bf16 TRANSPOSED [w][e][d].
// ---------------------------------------------------------------------------
__global__ __launch_bounds__(256) void cvt_w_kernel(const float* __restrict__ Wq,
                                                    const float* __restrict__ Wk,
                                                    const float* __restrict__ Wv,
                                                    __bf16* __restrict__ Wt) {
  int idx = blockIdx.x * blockDim.x + threadIdx.x;
  if (idx >= 3 * DIM * DIM) return;
  int w = idx >> 18;
  int r = idx & (DIM * DIM - 1);
  int e = r >> 9;
  int d = r & (DIM - 1);
  const float* W = (w == 0) ? Wq : ((w == 1) ? Wk : Wv);
  Wt[idx] = f2bf(W[d * DIM + e]);
}

// ---------------------------------------------------------------------------
// Kernel 3: QKV projection.  grid = (BS/16, 3), block = 256 (8 waves).
// Q, K stored row-major bf16 [m][e]; V stored transposed bf16 [b][e][s].
// ---------------------------------------------------------------------------
__global__ __launch_bounds__(256) void qkv_kernel(const __bf16* __restrict__ Xbf,
                                                  const __bf16* __restrict__ Wt,
                                                  const float* __restrict__ bq,
                                                  const float* __restrict__ bk,
                                                  const float* __restrict__ bv,
                                                  __bf16* __restrict__ Qbf,
                                                  __bf16* __restrict__ Kbf,
                                                  __bf16* __restrict__ Vt) {
  __shared__ __bf16 ldsA[16 * DIM];                 // 16 KB
  const int mt = blockIdx.x;
  const int w  = blockIdx.y;
  const int tid = threadIdx.x;
  const int wave = tid >> 5, lane = tid & 31;
  const int n = lane & 15, kg = lane >> 4;

  stage_tile_contig(ldsA, Xbf + (size_t)mt * 16 * DIM, 16 * DIM);
  __syncthreads();

  const __bf16* Wsel = Wt + (size_t)w * DIM * DIM;  // [e][d] transposed
  v8f acc[4];
  for (int t = 0; t < 4; ++t) acc[t] = v8f_zero();

  for (int ks = 0; ks < DIM / 32; ++ks) {
    v16bf a = load_a_frag(ldsA + ks * 32, DIM);
    for (int t = 0; t < 4; ++t) {
      int e0 = wave * 64 + t * 16;
      v16bf b = load_b_frag(Wsel + (size_t)e0 * DIM + ks * 32, DIM);
      acc[t] = wmma_bf16(a, b, acc[t]);
    }
  }

  const float* bsel = (w == 0) ? bq : ((w == 1) ? bk : bv);
  for (int t = 0; t < 4; ++t) {
    int e = wave * 64 + t * 16 + n;
    float bias = bsel[e];
    for (int r = 0; r < 8; ++r) {
      int m = mt * 16 + r + kg * 8;                  // global row (b*SEQ + s)
      __bf16 h = f2bf(acc[t][r] + bias);
      if (w == 2) {
        int bi = m >> 11, s = m & (SEQ - 1);
        Vt[(size_t)bi * DIM * SEQ + (size_t)e * SEQ + s] = h;
      } else if (w == 1) {
        Kbf[(size_t)m * DIM + e] = h;
      } else {
        Qbf[(size_t)m * DIM + e] = h;
      }
    }
  }
}

// ---------------------------------------------------------------------------
// Kernel 4: fused scores -> tanh -> softmax -> weights out -> context.
// grid = (SEQ/16, BATCH), block = 256 (8 waves).
// Dynamic LDS: ldsQ 16x512 bf16 (16KB) + ldsS 16x2048 f32 (128KB) = 144KB.
// Probs (bf16) round-trip through workspace (67MB total => L2-resident).
// ---------------------------------------------------------------------------
__global__ __launch_bounds__(256) void attn_kernel(const __bf16* __restrict__ Qbf,
                                                   const __bf16* __restrict__ Kbf,
                                                   const __bf16* __restrict__ Vt,
                                                   __bf16* __restrict__ Pbf,
                                                   float* __restrict__ ctx,
                                                   float* __restrict__ wts) {
  extern __shared__ char smem[];
  __bf16* ldsQ = (__bf16*)smem;                          // 16*512 bf16
  float*  ldsS = (float*)(smem + 16 * DIM * 2);          // 16*2048 f32

  const int qt = blockIdx.x;
  const int b  = blockIdx.y;
  const int tid = threadIdx.x;
  const int wave = tid >> 5, lane = tid & 31;
  const int n = lane & 15, kg = lane >> 4;
  const size_t qrow0 = (size_t)b * SEQ + (size_t)qt * 16;

  stage_tile_contig(ldsQ, Qbf + qrow0 * DIM, 16 * DIM);
  __syncthreads();

  // ---- phase 1: scores = tanh((Q K^T) * scale) into LDS ---------------------
  // Per wave: 256 key columns as 4 groups of 4 tiles; one A-frag feeds 4
  // independent WMMA accumulation chains (matches qkv scheduling shape).
  const __bf16* Kb = Kbf + (size_t)b * SEQ * DIM;        // [s][d] row-major
  const float scale = 0.04419417382415922f;              // 1/sqrt(512)
  for (int cg = 0; cg < 4; ++cg) {
    int col0 = wave * 256 + cg * 64;
    v8f acc[4];
    for (int t = 0; t < 4; ++t) acc[t] = v8f_zero();
    __builtin_prefetch(Kb + (size_t)(col0 + 64) * DIM, 0, 1);
    for (int ks = 0; ks < DIM / 32; ++ks) {
      v16bf a = load_a_frag(ldsQ + ks * 32, DIM);
      for (int t = 0; t < 4; ++t) {
        v16bf bb = load_b_frag(Kb + (size_t)(col0 + t * 16) * DIM + ks * 32, DIM);
        acc[t] = wmma_bf16(a, bb, acc[t]);
      }
    }
    for (int t = 0; t < 4; ++t)
      for (int r = 0; r < 8; ++r)
        ldsS[(r + kg * 8) * SEQ + col0 + t * 16 + n] = tanhf(acc[t][r] * scale);
  }
  __syncthreads();

  // ---- phase 2: per-row softmax over 2048, weights (f32) + probs (bf16) -----
  {
    int row = tid >> 4, li = tid & 15;                   // 16 threads per row
    float* Srow = ldsS + row * SEQ;
    float m = -1e30f;
    for (int j = li; j < SEQ; j += 16) m = fmaxf(m, Srow[j]);
    for (int off = 8; off; off >>= 1) m = fmaxf(m, __shfl_xor(m, off, 16));
    float sum = 0.f;
    for (int j = li; j < SEQ; j += 16) {
      float e = __expf(Srow[j] - m);
      Srow[j] = e;
      sum += e;
    }
    for (int off = 8; off; off >>= 1) sum += __shfl_xor(sum, off, 16);
    float inv = 1.f / sum;
    float*  wrow = wts + (qrow0 + row) * SEQ;
    __bf16* Prow = Pbf + ((size_t)(b * (SEQ / 16) + qt) * 16 + row) * SEQ;
    for (int j = li; j < SEQ; j += 16) {
      float p = Srow[j] * inv;
      wrow[j] = p;
      Prow[j] = f2bf(p);
    }
  }
  __threadfence();
  __syncthreads();

  // ---- phase 3: context = P @ V : one A-frag -> 4 WMMA chains ---------------
  const __bf16* Pblk = Pbf + (size_t)(b * (SEQ / 16) + qt) * 16 * SEQ;  // [16][2048]
  const __bf16* Vb   = Vt + (size_t)b * DIM * SEQ;                      // [e][s]
  {
    v8f acc[4];
    for (int t = 0; t < 4; ++t) acc[t] = v8f_zero();
    for (int kt = 0; kt < SEQ / 32; ++kt) {
      v16bf a = load_a_frag(Pblk + kt * 32, SEQ);
      __builtin_prefetch(Vb + (size_t)(wave * 64) * SEQ + (kt + 2) * 32, 0, 1);
      for (int t = 0; t < 4; ++t) {
        v16bf bb = load_b_frag(Vb + (size_t)(wave * 64 + t * 16) * SEQ + kt * 32, SEQ);
        acc[t] = wmma_bf16(a, bb, acc[t]);
      }
    }
    for (int t = 0; t < 4; ++t) {
      int e0 = wave * 64 + t * 16;
      for (int r = 0; r < 8; ++r)
        ctx[(qrow0 + r + kg * 8) * DIM + e0 + n] = acc[t][r];
    }
  }
}

// ---------------------------------------------------------------------------
// Host-side launcher
// ---------------------------------------------------------------------------
extern "C" void kernel_launch(void* const* d_in, const int* in_sizes, int n_in,
                              void* d_out, int out_size, void* d_ws, size_t ws_size,
                              hipStream_t stream) {
  const float* x  = (const float*)d_in[0];
  const float* Wq = (const float*)d_in[1];
  const float* bq = (const float*)d_in[2];
  const float* Wk = (const float*)d_in[3];
  const float* bk = (const float*)d_in[4];
  const float* Wv = (const float*)d_in[5];
  const float* bv = (const float*)d_in[6];

  float* ctx = (float*)d_out;                            // [8,2048,512]
  float* wts = ctx + (size_t)BATCH * SEQ * DIM;          // [8,2048,2048]

  char* ws = (char*)d_ws;
  __bf16* Wt  = (__bf16*)ws;                             // 3*512*512   (1.5 MiB)
  __bf16* Xbf = (__bf16*)(ws + (2ull << 20));            // 16384*512   (16 MiB)
  __bf16* Qbf = Xbf + (size_t)BS * DIM;                  // 16 MiB
  __bf16* Kbf = Qbf + (size_t)BS * DIM;                  // 16 MiB
  __bf16* Vt  = Kbf + (size_t)BS * DIM;                  // [b][e][s], 16 MiB
  __bf16* Pbf = Vt  + (size_t)BS * DIM;                  // [b*qt][16][2048], 64 MiB

  cvt_x_kernel<<<(BS * DIM / 8) / 256, 256, 0, stream>>>(x, Xbf);
  cvt_w_kernel<<<(3 * DIM * DIM) / 256, 256, 0, stream>>>(Wq, Wk, Wv, Wt);

  qkv_kernel<<<dim3(BS / 16, 3), 256, 0, stream>>>(Xbf, Wt, bq, bk, bv,
                                                   Qbf, Kbf, Vt);

  size_t smem = (size_t)16 * DIM * 2 + (size_t)16 * SEQ * 4;   // 144 KB dyn LDS
  attn_kernel<<<dim3(SEQ / 16, BATCH), 256, smem, stream>>>(Qbf, Kbf, Vt,
                                                            Pbf, ctx, wts);
}
